// InteracitveNeckLSS_10264971837638
// MI455X (gfx1250) — matrix operand order repaired and checked by reading, not compiled, not run on hardware
//
#include <hip/hip_runtime.h>
#include <cmath>

typedef _Float16 v16h __attribute__((ext_vector_type(16)));
typedef _Float16 v8h  __attribute__((ext_vector_type(8)));
typedef float    v8f  __attribute__((ext_vector_type(8)));

#define BNS 0.9999950000374997f   // 1/sqrt(1+1e-5)
#define P3  27648                 // 48*48*12 voxels

// ---------------- input indexing (setup_inputs dict order, params flattened) -
enum {
  IN_X = 0, IN_PROJ,
  P_DP1_W, P_DP1_B, P_DP1_G, P_DP1_BT,
  P_DP2_W, P_DP2_B,
  P_BLK_W, P_BLK_B, P_BLK_G, P_BLK_BT,
  P_ATTN_W, P_ATTN_B,
  P_N1_G, P_N1_BT,
  P_DS_W, P_DS_B, P_DS_G, P_DS_BT,
  P_PB1_W, P_PB1_B, P_PB1_G, P_PB1_BT,
  P_PB1_IW, P_PB1_IB, P_PB1_IG, P_PB1_IBT,
  P_PB2_W, P_PB2_B, P_PB2_G, P_PB2_BT,
  P_PBF_W, P_PBF_B, P_PBF_G, P_PBF_BT,
  P_PBF_IW, P_PBF_IB, P_PBF_IG, P_PBF_IBT
};

// ---------------- small helpers ---------------------------------------------
__global__ void zero_k(float* __restrict__ z, int n) {
  int i = blockIdx.x * blockDim.x + threadIdx.x;
  if (i < n) z[i] = 0.f;
}

__global__ void fuse_bn_k(const float* __restrict__ cb, const float* __restrict__ g,
                          const float* __restrict__ bt, float* __restrict__ sc,
                          float* __restrict__ bi, int C) {
  int i = blockIdx.x * blockDim.x + threadIdx.x;
  if (i < C) { float s = g[i] * BNS; sc[i] = s; bi[i] = cb[i] * s + bt[i]; }
}

__global__ void add_relu_k(const float* __restrict__ a, const float* __restrict__ b,
                           float* __restrict__ o, int n) {
  int i = blockIdx.x * blockDim.x + threadIdx.x;
  if (i < n) o[i] = fmaxf(a[i] + (b ? b[i] : 0.f), 0.f);
}

// NCHW camera features -> pixel-major [cam][h][w][64], f32 + f16 twins
__global__ void xprep_k(const float* __restrict__ x, float* __restrict__ xT,
                        _Float16* __restrict__ xTH) {
  int t = blockIdx.x * blockDim.x + threadIdx.x;
  if (t >= 6 * 7680 * 64) return;
  int c = t & 63, pq = t >> 6;
  int cam = pq / 7680, q = pq - cam * 7680;
  float v = x[(size_t)(cam * 64 + c) * 7680 + q];
  xT[t] = v;
  xTH[t] = (_Float16)v;
}

// w[Cout][Cin][T] (f32) -> wT[Cout][T][Cin] (f16)
__global__ void wprep_k(const float* __restrict__ w, _Float16* __restrict__ wT,
                        int Cout, int Cin, int T) {
  int t = blockIdx.x * blockDim.x + threadIdx.x;
  if (t >= Cout * T * Cin) return;
  int ci = t % Cin, r = t / Cin;
  int tap = r % T, oc = r / T;
  wT[t] = (_Float16)w[((size_t)oc * Cin + ci) * T + tap];
}

#define LOADH16(P) __builtin_shufflevector(*(const v8h*)(P), *(const v8h*)((P) + 8), \
                                           0,1,2,3,4,5,6,7,8,9,10,11,12,13,14,15)
#define LOADA16(P, KO) __builtin_shufflevector(*(const v8h*)((P) + (KO)), \
                                               *(const v8h*)((P) + (KO) + 16), \
                                               0,1,2,3,4,5,6,7,8,9,10,11,12,13,14,15)

// ---------------- implicit-GEMM 2D conv via WMMA (all-f16 operands) ---------
// in  : pixel-major f16 [Nimg*H*W][Cstride]  (reads channels 0..Cin-1)
// wgt : f16 [Cout][KH*KW][Cin];  out : pixel-major f32 [Nimg*H*W][Cout]
// wave computes 64(M) x 16(N); grid: x = Cout/64, y = Npix/64; block = 128.
template<int KH, int KW>
__global__ __launch_bounds__(128) void conv2d_wmma_k(
    const _Float16* __restrict__ in, const _Float16* __restrict__ wgt,
    const float* __restrict__ sc, const float* __restrict__ bi,
    const _Float16* __restrict__ zbuf, float* __restrict__ out,
    int Cin, int Cstride, int H, int W, int Cout)
{
  const int HW = H * W;
  const int T = KH * KW;
  const int Ktot = Cin * T;
  int lane = threadIdx.x & 31;
  int wave = threadIdx.x >> 5;
  int m0   = blockIdx.x * 64;
  int col  = (blockIdx.y * 4 + wave) * 16 + (lane & 15);
  int img  = col / HW;
  int q    = col - img * HW;
  int oy   = q / W, ox = q - oy * W;
  int kA   = (lane >> 4) * 8;
  int kB   = (lane >> 4) * 8;
  const _Float16* wr0 = wgt + (size_t)(m0      + (lane & 15)) * Ktot;
  const _Float16* wr1 = wgt + (size_t)(m0 + 16 + (lane & 15)) * Ktot;
  const _Float16* wr2 = wgt + (size_t)(m0 + 32 + (lane & 15)) * Ktot;
  const _Float16* wr3 = wgt + (size_t)(m0 + 48 + (lane & 15)) * Ktot;
  v8f a0 = {}, a1 = {}, a2 = {}, a3 = {};
  for (int tap = 0; tap < T; ++tap) {
    int ky = tap / KW, kx = tap - (tap / KW) * KW;
    int iy = oy + ky - KH / 2;
    int ix = ox + kx - KW / 2;
    bool ok = ((unsigned)iy < (unsigned)H) && ((unsigned)ix < (unsigned)W);
    const _Float16* bb = ok ? (in + ((size_t)(img * H + iy) * W + ix) * Cstride) : zbuf;
    int wo = tap * Cin;
    for (int c0 = 0; c0 < Cin; c0 += 32) {
      v16h b = LOADA16(bb + c0, kB);
      int ka = wo + c0 + kA;
      a0 = __builtin_amdgcn_wmma_f32_16x16x32_f16(false, LOADA16(wr0, ka), false, b, (short)0, a0, false, false);
      a1 = __builtin_amdgcn_wmma_f32_16x16x32_f16(false, LOADA16(wr1, ka), false, b, (short)0, a1, false, false);
      a2 = __builtin_amdgcn_wmma_f32_16x16x32_f16(false, LOADA16(wr2, ka), false, b, (short)0, a2, false, false);
      a3 = __builtin_amdgcn_wmma_f32_16x16x32_f16(false, LOADA16(wr3, ka), false, b, (short)0, a3, false, false);
    }
  }
  int mo = (lane >> 4) * 8;   // D: VGPR r -> row r (lanes>=16: +8)
  float* op = out + (size_t)col * Cout + m0 + mo;
  v8f accs[4] = {a0, a1, a2, a3};
#pragma unroll
  for (int mi = 0; mi < 4; ++mi) {
    int ocb = m0 + mi * 16 + mo;
    float4 s0 = *(const float4*)(sc + ocb), s1 = *(const float4*)(sc + ocb + 4);
    float4 c0 = *(const float4*)(bi + ocb), c1 = *(const float4*)(bi + ocb + 4);
    float4 o0, o1;
    o0.x = accs[mi][0]*s0.x + c0.x;  o0.y = accs[mi][1]*s0.y + c0.y;
    o0.z = accs[mi][2]*s0.z + c0.z;  o0.w = accs[mi][3]*s0.w + c0.w;
    o1.x = accs[mi][4]*s1.x + c1.x;  o1.y = accs[mi][5]*s1.y + c1.y;
    o1.z = accs[mi][6]*s1.z + c1.z;  o1.w = accs[mi][7]*s1.w + c1.w;
    *(float4*)(op + mi * 16)     = o0;
    *(float4*)(op + mi * 16 + 4) = o1;
  }
}

// ---------------- implicit-GEMM 3D conv (3x3x3, pad 1, z-stride SZ) ---------
template<int SZ>
__global__ __launch_bounds__(128) void conv3d_wmma_k(
    const _Float16* __restrict__ in, const _Float16* __restrict__ wgt,
    const float* __restrict__ sc, const float* __restrict__ bi,
    const _Float16* __restrict__ zbuf, float* __restrict__ out,
    int Cin, int Cout, int Zin, int Zo)
{
  const int Ktot = Cin * 27;
  int lane = threadIdx.x & 31;
  int wave = threadIdx.x >> 5;
  int m0   = blockIdx.x * 64;
  int col  = (blockIdx.y * 4 + wave) * 16 + (lane & 15);
  int vx = col / (48 * Zo);
  int r0 = col - vx * (48 * Zo);
  int vy = r0 / Zo;
  int vz = r0 - vy * Zo;
  int kA = (lane >> 4) * 8;
  int kB = (lane >> 4) * 8;
  const _Float16* wr0 = wgt + (size_t)(m0      + (lane & 15)) * Ktot;
  const _Float16* wr1 = wgt + (size_t)(m0 + 16 + (lane & 15)) * Ktot;
  const _Float16* wr2 = wgt + (size_t)(m0 + 32 + (lane & 15)) * Ktot;
  const _Float16* wr3 = wgt + (size_t)(m0 + 48 + (lane & 15)) * Ktot;
  v8f a0 = {}, a1 = {}, a2 = {}, a3 = {};
  for (int tap = 0; tap < 27; ++tap) {
    int tx = tap / 9, r2 = tap - tx * 9;
    int ty = r2 / 3, tz = r2 - ty * 3;
    int ixx = vx + tx - 1;
    int iyy = vy + ty - 1;
    int izz = vz * SZ + tz - 1;
    bool ok = ((unsigned)ixx < 48u) && ((unsigned)iyy < 48u) &&
              ((unsigned)izz < (unsigned)Zin);
    const _Float16* bb = ok ? (in + ((size_t)(ixx * 48 + iyy) * Zin + izz) * Cin) : zbuf;
    int wo = tap * Cin;
    for (int c0 = 0; c0 < Cin; c0 += 32) {
      v16h b = LOADA16(bb + c0, kB);
      int ka = wo + c0 + kA;
      a0 = __builtin_amdgcn_wmma_f32_16x16x32_f16(false, LOADA16(wr0, ka), false, b, (short)0, a0, false, false);
      a1 = __builtin_amdgcn_wmma_f32_16x16x32_f16(false, LOADA16(wr1, ka), false, b, (short)0, a1, false, false);
      a2 = __builtin_amdgcn_wmma_f32_16x16x32_f16(false, LOADA16(wr2, ka), false, b, (short)0, a2, false, false);
      a3 = __builtin_amdgcn_wmma_f32_16x16x32_f16(false, LOADA16(wr3, ka), false, b, (short)0, a3, false, false);
    }
  }
  int mo = (lane >> 4) * 8;
  float* op = out + (size_t)col * Cout + m0 + mo;
  v8f accs[4] = {a0, a1, a2, a3};
#pragma unroll
  for (int mi = 0; mi < 4; ++mi) {
    int ocb = m0 + mi * 16 + mo;
    float4 s0 = *(const float4*)(sc + ocb), s1 = *(const float4*)(sc + ocb + 4);
    float4 c0 = *(const float4*)(bi + ocb), c1 = *(const float4*)(bi + ocb + 4);
    float4 o0, o1;
    o0.x = accs[mi][0]*s0.x + c0.x;  o0.y = accs[mi][1]*s0.y + c0.y;
    o0.z = accs[mi][2]*s0.z + c0.z;  o0.w = accs[mi][3]*s0.w + c0.w;
    o1.x = accs[mi][4]*s1.x + c1.x;  o1.y = accs[mi][5]*s1.y + c1.y;
    o1.z = accs[mi][6]*s1.z + c1.z;  o1.w = accs[mi][7]*s1.w + c1.w;
    *(float4*)(op + mi * 16)     = o0;
    *(float4*)(op + mi * 16 + 4) = o1;
  }
}

// ---------------- dp2 1x1 conv (128->20) + channel softmax ------------------
__global__ void dp2_softmax_k(const float* __restrict__ h, const float* __restrict__ w,
                              const float* __restrict__ b, float* __restrict__ dp)
{
  int t = blockIdx.x * blockDim.x + threadIdx.x;
  if (t >= 6 * 7680) return;
  const float* hp = h + (size_t)t * 128;
  float v[20];
#pragma unroll
  for (int d = 0; d < 20; ++d) v[d] = b[d];
  for (int c = 0; c < 128; ++c) {
    float hv = hp[c];
#pragma unroll
    for (int d = 0; d < 20; ++d) v[d] += w[d * 128 + c] * hv;
  }
  float mx = v[0];
#pragma unroll
  for (int d = 1; d < 20; ++d) mx = fmaxf(mx, v[d]);
  float s = 0.f;
#pragma unroll
  for (int d = 0; d < 20; ++d) { v[d] = expf(v[d] - mx); s += v[d]; }
  float inv = 1.f / s;
#pragma unroll
  for (int d = 0; d < 20; ++d) dp[(size_t)t * 20 + d] = v[d] * inv;
}

// ---------------- voxel -> image projection ---------------------------------
__global__ void project_k(const float* __restrict__ proj, float* __restrict__ ixb,
                          float* __restrict__ iyb, float* __restrict__ izb,
                          float* __restrict__ cnt)
{
  int p = blockIdx.x * blockDim.x + threadIdx.x;
  if (p >= P3) return;
  int i = p / 576, rem = p - i * 576;
  int j = rem / 12, k = rem - j * 12;
  float X = i * 0.8f  - 19.2f;
  float Y = j * 0.8f  - 19.2f;
  float Z = k * 0.32f - 1.92f;
  float c = 0.f;
  for (int cam = 0; cam < 6; ++cam) {
    const float* Pm = proj + cam * 12;
    float p0 = Pm[0] * X + Pm[1] * Y + Pm[2]  * Z + Pm[3];
    float p1 = Pm[4] * X + Pm[5] * Y + Pm[6]  * Z + Pm[7];
    float p2 = Pm[8] * X + Pm[9] * Y + Pm[10] * Z + Pm[11];
    float px = p0 / (p2 + 1e-10f);
    float py = p1 / (p2 + 1e-10f);
    bool valid = (px >= 0.f) && (py >= 0.f) && (px < 160.f) && (py < 48.f) && (p2 > 0.f);
    c += valid ? 1.f : 0.f;
    ixb[cam * P3 + p] = px * (159.f / 160.f);
    iyb[cam * P3 + p] = py * (47.f  / 48.f);
    izb[cam * P3 + p] = p2 * (19.f  / 60.f);
  }
  cnt[p] = c;
}

// ------- trilinear sample of outer=dp*x; thread = voxel*64 + channel --------
__global__ void vol3_k(const float* __restrict__ xT, const float* __restrict__ dpT,
                       const float* __restrict__ ixb, const float* __restrict__ iyb,
                       const float* __restrict__ izb, const float* __restrict__ cnt,
                       float* __restrict__ vox, _Float16* __restrict__ voxh)
{
  int t = blockIdx.x * blockDim.x + threadIdx.x;
  if (t >= 64 * P3) return;
  int c = t & 63, p = t >> 6;           // wave-uniform voxel, lane = channel
  float acc = 0.f;
  for (int cam = 0; cam < 6; ++cam) {
    float ix = ixb[cam * P3 + p], iy = iyb[cam * P3 + p], iz = izb[cam * P3 + p];
    float x0f = floorf(ix), y0f = floorf(iy), z0f = floorf(iz);
    int x0 = (int)x0f, y0 = (int)y0f, z0 = (int)z0f;
    float fx = ix - x0f, fy = iy - y0f, fz = iz - z0f;
    for (int dy = 0; dy < 2; ++dy) {
      int yi = y0 + dy; if ((unsigned)yi >= 48u) continue;
      float wy = dy ? fy : 1.f - fy;
      for (int dx = 0; dx < 2; ++dx) {
        int xi = x0 + dx; if ((unsigned)xi >= 160u) continue;
        float wx = dx ? fx : 1.f - fx;
        size_t pix = (size_t)(cam * 48 + yi) * 160 + xi;
        float dpa = 0.f;
        for (int dz = 0; dz < 2; ++dz) {
          int zi = z0 + dz; if ((unsigned)zi >= 20u) continue;
          float wz = dz ? fz : 1.f - fz;
          dpa += wz * dpT[pix * 20 + zi];
        }
        acc += wx * wy * dpa * xT[pix * 64 + c];
      }
    }
  }
  float cv = cnt[p];
  float v = (cv != 0.f) ? acc / (cv + 1e-10f) : 0.f;
  vox[t] = v;
  voxh[t] = (_Float16)v;
}

// ------ unfold-3x3 bilinear sample + depthwise attention + n1-BN + residual -
// writes f16 only (sole consumer is the ds WMMA conv)
__global__ void vu_attn_k(const float* __restrict__ xT, const float* __restrict__ ixb,
                          const float* __restrict__ iyb, const float* __restrict__ cnt,
                          const float* __restrict__ outb, const float* __restrict__ aw,
                          const float* __restrict__ ab, const float* __restrict__ n1g,
                          const float* __restrict__ n1bt, _Float16* __restrict__ o)
{
  int t = blockIdx.x * blockDim.x + threadIdx.x;
  if (t >= 64 * P3) return;
  int c = t & 63, p = t >> 6;
  float cv  = cnt[p];
  float inv = (cv != 0.f) ? 1.f / (cv + 1e-10f) : 0.f;
  int vx = p / 576, rem = p - vx * 576;
  int vy = rem / 12, vz = rem - vy * 12;
  float acc = 0.f;
  for (int tap = 0; tap < 9; ++tap) {
    int ti = tap / 3 - 1, tj = tap % 3 - 1;
    float vu = 0.f;
    for (int cam = 0; cam < 6; ++cam) {
      float ix = ixb[cam * P3 + p], iy = iyb[cam * P3 + p];
      float x0f = floorf(ix), y0f = floorf(iy);
      int x0 = (int)x0f, y0 = (int)y0f;
      float fx = ix - x0f, fy = iy - y0f;
      for (int dy = 0; dy < 2; ++dy) {
        int yi = y0 + dy; if ((unsigned)yi >= 48u) continue;
        float wy = dy ? fy : 1.f - fy;
        for (int dx = 0; dx < 2; ++dx) {
          int xi = x0 + dx; if ((unsigned)xi >= 160u) continue;
          float wx = dx ? fx : 1.f - fx;
          int yy = yi + ti, xx = xi + tj;      // zero pad of unfold
          float v = ((unsigned)yy < 48u && (unsigned)xx < 160u)
                    ? xT[((size_t)(cam * 48 + yy) * 160 + xx) * 64 + c] : 0.f;
          vu += wx * wy * v;
        }
      }
    }
    vu *= inv;
    int oc = c * 9 + tap;
    float a = ab[oc];
    const float* wb = aw + (size_t)oc * 27;
    for (int tx = 0; tx < 3; ++tx) {
      int nx = vx + tx - 1; if ((unsigned)nx >= 48u) continue;
      for (int ty = 0; ty < 3; ++ty) {
        int ny = vy + ty - 1; if ((unsigned)ny >= 48u) continue;
        for (int tz = 0; tz < 3; ++tz) {
          int nz = vz + tz - 1; if ((unsigned)nz >= 12u) continue;
          a += wb[tx * 9 + ty * 3 + tz] *
               outb[((size_t)(nx * 576 + ny * 12 + nz)) * 64 + c];
        }
      }
    }
    float s = 1.f / (1.f + expf(-a));
    acc += vu * s;
  }
  float y = acc * (n1g[c] * BNS) + n1bt[c] + outb[t];
  o[t] = (_Float16)fmaxf(y, 0.f);
}

// -------- bev = mean over z; writes channels [0,128) of the f16 concat ------
__global__ void bev_mean_k(const float* __restrict__ v, _Float16* __restrict__ cat) {
  int t = blockIdx.x * blockDim.x + threadIdx.x;
  if (t >= 2304 * 128) return;
  int c = t & 127, pix = t >> 7;
  float s = 0.f;
#pragma unroll
  for (int z = 0; z < 6; ++z) s += v[((size_t)pix * 6 + z) * 128 + c];
  cat[(size_t)pix * 640 + c] = (_Float16)(s * (1.f / 6.f));
}

// ---------------- 2x2 max pool (pixel-major), optional f16 twin -------------
__global__ void maxpool2_k(const float* __restrict__ in, float* __restrict__ out,
                           _Float16* __restrict__ outh, int C, int H, int W) {
  int Ho = H / 2, Wo = W / 2;
  int t = blockIdx.x * blockDim.x + threadIdx.x;
  if (t >= C * Ho * Wo) return;
  int c = t % C, opix = t / C;
  int oy = opix / Wo, ox = opix - oy * Wo;
  const float* b = in + ((size_t)(2 * oy) * W + 2 * ox) * C + c;
  float m = fmaxf(b[0], b[C]);
  m = fmaxf(m, fmaxf(b[(size_t)W * C], b[(size_t)W * C + C]));
  out[t] = m;
  if (outh) outh[t] = (_Float16)m;
}

// ---------------- bilinear resize into strided f16 concat buffer ------------
__global__ void resize_k(const float* __restrict__ in, _Float16* __restrict__ out,
                         int C, int H, int W, int Ho, int Wo, int ostride, int ooff) {
  int t = blockIdx.x * blockDim.x + threadIdx.x;
  if (t >= C * Ho * Wo) return;
  int c = t % C, opix = t / C;
  int yo = opix / Wo, xo = opix - yo * Wo;
  float ys = yo * (float)(H - 1) / (float)(Ho - 1);
  float xs = xo * (float)(W - 1) / (float)(Wo - 1);
  int y0 = (int)floorf(ys), x0 = (int)floorf(xs);
  int y1 = min(y0 + 1, H - 1), x1 = min(x0 + 1, W - 1);
  float wy = ys - (float)y0, wx = xs - (float)x0;
  float t0 = in[((size_t)y0 * W + x0) * C + c] * (1.f - wy) +
             in[((size_t)y1 * W + x0) * C + c] * wy;
  float t1 = in[((size_t)y0 * W + x1) * C + c] * (1.f - wy) +
             in[((size_t)y1 * W + x1) * C + c] * wy;
  out[(size_t)opix * ostride + ooff + c] = (_Float16)(t0 * (1.f - wx) + t1 * wx);
}

// ---------------- outputs ---------------------------------------------------
__global__ void transpose_out_k(const float* __restrict__ f, float* __restrict__ o) {
  int t = blockIdx.x * blockDim.x + threadIdx.x;
  if (t >= 256 * 2304) return;
  int c = t / 2304, r = t - c * 2304;
  int yo = r / 48, xo = r - yo * 48;
  o[t] = f[((size_t)(xo * 48 + yo)) * 256 + c];
}

__global__ void valids_out_k(const float* __restrict__ cnt, float* __restrict__ o) {
  int p = blockIdx.x * blockDim.x + threadIdx.x;
  if (p >= P3) return;
  int vx = p / 576, rem = p - vx * 576;
  int vy = rem / 12, vz = rem - vy * 12;
  o[vx * 576 + vz * 48 + vy] = cnt[p];          // (X,Y,Z) -> (X,Z,Y)
}

// ---------------- workspace layout (float slots, all offsets 64B-aligned) ---
static const size_t OFF_XT    = 0;                          // 6*7680*64 f32
static const size_t OFF_XTH   = OFF_XT    + 2949120;        // same, f16 (1474560 slots)
static const size_t OFF_H     = OFF_XTH   + 1474560;        // 46080*128 f32
static const size_t OFF_DP    = OFF_H     + 5898240;        // 46080*20 f32
static const size_t OFF_IX    = OFF_DP    + 921600;         // 6*P3
static const size_t OFF_IY    = OFF_IX    + 165888;
static const size_t OFF_IZ    = OFF_IY    + 165888;
static const size_t OFF_CNT   = OFF_IZ    + 165888;         // P3
static const size_t OFF_VOX   = OFF_CNT   + 27648;          // P3*64 f32
static const size_t OFF_VOXH  = OFF_VOX   + 1769472;        // P3*64 f16 (884736)
static const size_t OFF_OUTB  = OFF_VOXH  + 884736;         // P3*64 f32
static const size_t OFF_VOX2H = OFF_OUTB  + 1769472;        // P3*64 f16 (884736)
static const size_t OFF_VOX3  = OFF_VOX2H + 884736;         // 13824*128 f32
static const size_t OFF_CATH  = OFF_VOX3  + 1769472;        // 2304*640 f16 (737280)
static const size_t OFF_Y1    = OFF_CATH  + 737280;         // 2304*256 f32
static const size_t OFF_R1    = OFF_Y1    + 589824;         // 2304*256 f32
static const size_t OFF_X1    = OFF_R1    + 589824;         // 576*256 f32
static const size_t OFF_X1H   = OFF_X1    + 147456;         // 576*256 f16 (73728)
static const size_t OFF_Y2    = OFF_X1H   + 73728;          // 576*256 f32
static const size_t OFF_X2    = OFF_Y2    + 147456;         // 144*256 f32
static const size_t OFF_YF    = OFF_X2    + 36864;          // 2304*256 f32
static const size_t OFF_RF    = OFF_YF    + 589824;         // 2304*256 f32
static const size_t OFF_SC    = OFF_RF    + 589824;         // 1024
static const size_t OFF_BI    = OFF_SC    + 1024;           // 1024
static const size_t OFF_Z     = OFF_BI    + 1024;           // 1024 zeros (f32/f16 view)
static const size_t OFF_WGT   = OFF_Z     + 1024;           // 1474560 f16 (737280)
// total ~ 23.1M float slots ~ 92 MB

extern "C" void kernel_launch(void* const* d_in, const int* in_sizes, int n_in,
                              void* d_out, int out_size, void* d_ws, size_t ws_size,
                              hipStream_t stream)
{
  (void)in_sizes; (void)n_in; (void)out_size; (void)ws_size;
  const float* x    = (const float*)d_in[IN_X];
  const float* proj = (const float*)d_in[IN_PROJ];
  auto F = [&](int i) { return (const float*)d_in[i]; };

  float* ws    = (float*)d_ws;
  float* XT    = ws + OFF_XT;
  _Float16* XTH   = (_Float16*)(ws + OFF_XTH);
  float* Hbuf  = ws + OFF_H;
  float* DP    = ws + OFF_DP;
  float* IX    = ws + OFF_IX;
  float* IY    = ws + OFF_IY;
  float* IZ    = ws + OFF_IZ;
  float* CNT   = ws + OFF_CNT;
  float* VOX   = ws + OFF_VOX;
  _Float16* VOXH  = (_Float16*)(ws + OFF_VOXH);
  float* OUTB  = ws + OFF_OUTB;
  _Float16* VOX2H = (_Float16*)(ws + OFF_VOX2H);
  float* VOX3  = ws + OFF_VOX3;
  _Float16* CATH  = (_Float16*)(ws + OFF_CATH);
  float* Y1    = ws + OFF_Y1;
  float* R1    = ws + OFF_R1;
  float* X1    = ws + OFF_X1;
  _Float16* X1H   = (_Float16*)(ws + OFF_X1H);
  float* Y2    = ws + OFF_Y2;
  float* X2    = ws + OFF_X2;
  float* YF    = ws + OFF_YF;
  float* RF    = ws + OFF_RF;
  float* SC    = ws + OFF_SC;
  float* BI    = ws + OFF_BI;
  float* ZB    = ws + OFF_Z;
  const _Float16* ZBH = (const _Float16*)ZB;   // zero bits are zero in f16 too
  _Float16* WGT = (_Float16*)(ws + OFF_WGT);

  zero_k<<<4, 256, 0, stream>>>(ZB, 1024);
  xprep_k<<<(2949120 + 255) / 256, 256, 0, stream>>>(x, XT, XTH);

  // dp1: 3x3 conv 64->128 + BN + ReLU   (M=128, K=576, N=46080)
  fuse_bn_k<<<1, 256, 0, stream>>>(F(P_DP1_B), F(P_DP1_G), F(P_DP1_BT), SC, BI, 128);
  wprep_k<<<(73728 + 255) / 256, 256, 0, stream>>>(F(P_DP1_W), WGT, 128, 64, 9);
  conv2d_wmma_k<3, 3><<<dim3(2, 720), 128, 0, stream>>>(XTH, WGT, SC, BI, ZBH, Hbuf,
                                                        64, 64, 48, 160, 128);
  add_relu_k<<<(5898240 + 255) / 256, 256, 0, stream>>>(Hbuf, nullptr, Hbuf, 5898240);

  // dp2 + softmax
  dp2_softmax_k<<<(46080 + 255) / 256, 256, 0, stream>>>(Hbuf, F(P_DP2_W), F(P_DP2_B), DP);

  // voxel projection
  project_k<<<(P3 + 255) / 256, 256, 0, stream>>>(proj, IX, IY, IZ, CNT);

  // trilinear sample of dp*x, masked average (f32 + f16 twins)
  vol3_k<<<(64 * P3 + 255) / 256, 256, 0, stream>>>(XT, DP, IX, IY, IZ, CNT, VOX, VOXH);

  // blk: 3x3x3 conv 64->64 + BN, ReLU(+voxel)   (M=64, K=1728, N=27648)
  fuse_bn_k<<<1, 256, 0, stream>>>(F(P_BLK_B), F(P_BLK_G), F(P_BLK_BT), SC, BI, 64);
  wprep_k<<<(110592 + 255) / 256, 256, 0, stream>>>(F(P_BLK_W), WGT, 64, 64, 27);
  conv3d_wmma_k<1><<<dim3(1, 432), 128, 0, stream>>>(VOXH, WGT, SC, BI, ZBH, OUTB,
                                                     64, 64, 12, 12);
  add_relu_k<<<(64 * P3 + 255) / 256, 256, 0, stream>>>(OUTB, VOX, OUTB, 64 * P3);

  // unfold-sample * sigmoid(depthwise attn) + n1 BN + residual + ReLU -> f16
  vu_attn_k<<<(64 * P3 + 255) / 256, 256, 0, stream>>>(XT, IX, IY, CNT, OUTB,
                                                       F(P_ATTN_W), F(P_ATTN_B),
                                                       F(P_N1_G), F(P_N1_BT), VOX2H);

  // ds: 3x3x3 conv 64->128, z-stride 2, BN, ReLU   (M=128, K=1728, N=13824)
  fuse_bn_k<<<1, 256, 0, stream>>>(F(P_DS_B), F(P_DS_G), F(P_DS_BT), SC, BI, 128);
  wprep_k<<<(221184 + 255) / 256, 256, 0, stream>>>(F(P_DS_W), WGT, 128, 64, 27);
  conv3d_wmma_k<2><<<dim3(2, 216), 128, 0, stream>>>(VOX2H, WGT, SC, BI, ZBH, VOX3,
                                                     64, 128, 12, 6);
  add_relu_k<<<(128 * 13824 + 255) / 256, 256, 0, stream>>>(VOX3, nullptr, VOX3, 128 * 13824);

  // bev = mean over z (into f16 concat channels 0..127)
  bev_mean_k<<<(2304 * 128 + 255) / 256, 256, 0, stream>>>(VOX3, CATH);

  // pb1: 3x3 128->256 + 1x1 shortcut, ReLU, maxpool
  fuse_bn_k<<<1, 256, 0, stream>>>(F(P_PB1_B), F(P_PB1_G), F(P_PB1_BT), SC, BI, 256);
  wprep_k<<<(294912 + 255) / 256, 256, 0, stream>>>(F(P_PB1_W), WGT, 256, 128, 9);
  conv2d_wmma_k<3, 3><<<dim3(4, 36), 128, 0, stream>>>(CATH, WGT, SC, BI, ZBH, Y1,
                                                       128, 640, 48, 48, 256);
  fuse_bn_k<<<1, 256, 0, stream>>>(F(P_PB1_IB), F(P_PB1_IG), F(P_PB1_IBT), SC, BI, 256);
  wprep_k<<<(32768 + 255) / 256, 256, 0, stream>>>(F(P_PB1_IW), WGT, 256, 128, 1);
  conv2d_wmma_k<1, 1><<<dim3(4, 36), 128, 0, stream>>>(CATH, WGT, SC, BI, ZBH, R1,
                                                       128, 640, 48, 48, 256);
  add_relu_k<<<(589824 + 255) / 256, 256, 0, stream>>>(Y1, R1, Y1, 589824);
  maxpool2_k<<<(147456 + 255) / 256, 256, 0, stream>>>(Y1, X1, X1H, 256, 48, 48);

  // pb2: 3x3 256->256, identity residual, ReLU, maxpool
  fuse_bn_k<<<1, 256, 0, stream>>>(F(P_PB2_B), F(P_PB2_G), F(P_PB2_BT), SC, BI, 256);
  wprep_k<<<(589824 + 255) / 256, 256, 0, stream>>>(F(P_PB2_W), WGT, 256, 256, 9);
  conv2d_wmma_k<3, 3><<<dim3(4, 9), 128, 0, stream>>>(X1H, WGT, SC, BI, ZBH, Y2,
                                                      256, 256, 24, 24, 256);
  add_relu_k<<<(147456 + 255) / 256, 256, 0, stream>>>(Y2, X1, Y2, 147456);
  maxpool2_k<<<(36864 + 255) / 256, 256, 0, stream>>>(Y2, X2, nullptr, 256, 24, 24);

  // upsample into f16 concat channels [128,384) and [384,640)
  resize_k<<<(589824 + 255) / 256, 256, 0, stream>>>(X1, CATH, 256, 24, 24, 48, 48, 640, 128);
  resize_k<<<(589824 + 255) / 256, 256, 0, stream>>>(X2, CATH, 256, 12, 12, 48, 48, 640, 384);

  // pbf: 3x3 640->256 + 1x1 shortcut, ReLU   (M=256, K=5760, N=2304)
  fuse_bn_k<<<1, 256, 0, stream>>>(F(P_PBF_B), F(P_PBF_G), F(P_PBF_BT), SC, BI, 256);
  wprep_k<<<(1474560 + 255) / 256, 256, 0, stream>>>(F(P_PBF_W), WGT, 256, 640, 9);
  conv2d_wmma_k<3, 3><<<dim3(4, 36), 128, 0, stream>>>(CATH, WGT, SC, BI, ZBH, YF,
                                                       640, 640, 48, 48, 256);
  fuse_bn_k<<<1, 256, 0, stream>>>(F(P_PBF_IB), F(P_PBF_IG), F(P_PBF_IBT), SC, BI, 256);
  wprep_k<<<(163840 + 255) / 256, 256, 0, stream>>>(F(P_PBF_IW), WGT, 256, 640, 1);
  conv2d_wmma_k<1, 1><<<dim3(4, 36), 128, 0, stream>>>(CATH, WGT, SC, BI, ZBH, RF,
                                                       640, 640, 48, 48, 256);
  add_relu_k<<<(589824 + 255) / 256, 256, 0, stream>>>(YF, RF, YF, 589824);

  // outputs
  float* outp = (float*)d_out;
  transpose_out_k<<<(589824 + 255) / 256, 256, 0, stream>>>(YF, outp);
  valids_out_k<<<(P3 + 255) / 256, 256, 0, stream>>>(CNT, outp + 589824);
}
